// EntityPredictionHead_CandidateList_29094108463767
// MI455X (gfx1250) — matrix operand order repaired and checked by reading, not compiled, not run on hardware
//
#include <hip/hip_runtime.h>
#include <hip/hip_bf16.h>
#include <math.h>

#define HIDDEN 1024
#define EMB    256
#define NB     4096
#define NC     128
#define LN_EPS 1e-12f

typedef __attribute__((ext_vector_type(16))) __bf16 v16bf;
typedef __attribute__((ext_vector_type(8)))  float  v8f;

__device__ __forceinline__ v8f wmma_bf16(v16bf a, v16bf b, v8f c) {
    // (neg_a, A, neg_b, B, c_mod, C, reuse_a, reuse_b)
    return __builtin_amdgcn_wmma_f32_16x16x32_bf16(false, a, false, b, (short)0, c, false, false);
}

__device__ __forceinline__ float wave_reduce_sum(float v) {
#pragma unroll
    for (int off = 16; off > 0; off >>= 1)
        v += __shfl_xor(v, off, 32);
    return v;
}

__device__ __forceinline__ float gelu_exact(float x) {
    return 0.5f * x * (1.0f + erff(x * 0.70710678118654752440f));
}

// Split an f32 into bf16 hi + bf16 lo so that hi+lo ~= x with ~16 mantissa bits.
__device__ __forceinline__ void split_bf16(float x, __bf16& hi, __bf16& lo) {
    __bf16 h = (__bf16)x;
    hi = h;
    lo = (__bf16)(x - (float)h);
}

__global__ __launch_bounds__(256)
void entity_head_fused_kernel(const float* __restrict__ X,      // [NB, HIDDEN]
                              const float* __restrict__ W,      // [EMB, HIDDEN]
                              const float* __restrict__ bdense, // [EMB]
                              const float* __restrict__ gamma,  // [EMB]
                              const float* __restrict__ beta,   // [EMB]
                              const float* __restrict__ table,  // [VOCAB, EMB]
                              const float* __restrict__ ebias,  // [VOCAB]
                              const int*   __restrict__ cidx,   // [NB, NC]
                              float*       __restrict__ out)    // [NB, NC]
{
    __shared__ float hrow[16][EMB];   // 16 KB: gelu'd then normalized h tile

    const int tid  = threadIdx.x;
    const int wave = tid >> 5;        // 0..7
    const int lane = tid & 31;
    const int half = lane >> 4;       // 0: lanes 0-15, 1: lanes 16-31
    const int mn   = lane & 15;       // M (A,C/D rows-in-half) or N (B,C/D col)
    const int m0   = blockIdx.x * 16; // 16 batch rows per block

    // ---------------- Phase 1: h_pre = X @ W^T via bf16-split WMMA ----------
    const int n0 = wave * 32;         // each wave: output cols [n0, n0+32)
    v8f acc0 = {};
    v8f acc1 = {};
    const float* xrow  = X + (size_t)(m0 + mn) * HIDDEN;
    const float* wrow0 = W + (size_t)(n0 + mn)      * HIDDEN;
    const float* wrow1 = W + (size_t)(n0 + 16 + mn) * HIDDEN;

    for (int k0 = 0; k0 < HIDDEN; k0 += 32) {
        // A tile 16x32: lanes 0-15 hold K = j; lanes 16-31 hold K = 16 + j
        float af[16];
        {
            const float4 a0 = *(const float4*)(xrow + k0 + 16 * half);
            const float4 a1 = *(const float4*)(xrow + k0 + 16 * half + 4);
            const float4 a2 = *(const float4*)(xrow + k0 + 16 * half + 8);
            const float4 a3 = *(const float4*)(xrow + k0 + 16 * half + 12);
            af[0]=a0.x;  af[1]=a0.y;  af[2]=a0.z;  af[3]=a0.w;
            af[4]=a1.x;  af[5]=a1.y;  af[6]=a1.z;  af[7]=a1.w;
            af[8]=a2.x;  af[9]=a2.y;  af[10]=a2.z; af[11]=a2.w;
            af[12]=a3.x; af[13]=a3.y; af[14]=a3.z; af[15]=a3.w;
        }
        v16bf ahi, alo;
#pragma unroll
        for (int j = 0; j < 16; ++j) { __bf16 h, l; split_bf16(af[j], h, l); ahi[j] = h; alo[j] = l; }

        // B tiles 32x16 (B[k][n] = W[n][k]): lanes 0-15 K=k0..k0+15, lanes 16-31 K=k0+16..k0+31
        float bf0[16], bf1[16];
        {
            const float* p0 = wrow0 + k0 + 16 * half;
            const float* p1 = wrow1 + k0 + 16 * half;
#pragma unroll
            for (int q = 0; q < 4; ++q) {
                const float4 t0 = *(const float4*)(p0 + 4 * q);
                const float4 t1 = *(const float4*)(p1 + 4 * q);
                bf0[4*q+0]=t0.x; bf0[4*q+1]=t0.y; bf0[4*q+2]=t0.z; bf0[4*q+3]=t0.w;
                bf1[4*q+0]=t1.x; bf1[4*q+1]=t1.y; bf1[4*q+2]=t1.z; bf1[4*q+3]=t1.w;
            }
        }
        v16bf b0hi, b0lo, b1hi, b1lo;
#pragma unroll
        for (int j = 0; j < 16; ++j) {
            __bf16 h, l;
            split_bf16(bf0[j], h, l); b0hi[j] = h; b0lo[j] = l;
            split_bf16(bf1[j], h, l); b1hi[j] = h; b1lo[j] = l;
        }

        // 3-term split product: hi*hi + hi*lo + lo*hi (error ~2^-16 per element)
        acc0 = wmma_bf16(ahi, b0hi, acc0);
        acc0 = wmma_bf16(ahi, b0lo, acc0);
        acc0 = wmma_bf16(alo, b0hi, acc0);
        acc1 = wmma_bf16(ahi, b1hi, acc1);
        acc1 = wmma_bf16(ahi, b1lo, acc1);
        acc1 = wmma_bf16(alo, b1hi, acc1);
    }

    // bias + exact gelu, stash tile in LDS. D VGPR r holds (M = r + 8*half, N = mn).
#pragma unroll
    for (int r = 0; r < 8; ++r) {
        const int row = r + 8 * half;
        {
            const int col = n0 + mn;
            hrow[row][col] = gelu_exact(acc0[r] + bdense[col]);
        }
        {
            const int col = n0 + 16 + mn;
            hrow[row][col] = gelu_exact(acc1[r] + bdense[col]);
        }
    }
    __syncthreads();

    // ---------------- LayerNorm over EMB per row (2 rows per wave) ----------
#pragma unroll
    for (int rr = 0; rr < 2; ++rr) {
        const int row = wave * 2 + rr;
        float v[8];
        float s = 0.0f, s2 = 0.0f;
#pragma unroll
        for (int j = 0; j < 8; ++j) {
            v[j] = hrow[row][lane * 8 + j];
            s  += v[j];
            s2 += v[j] * v[j];
        }
        s  = wave_reduce_sum(s);
        s2 = wave_reduce_sum(s2);
        const float mu  = s * (1.0f / EMB);
        const float var = s2 * (1.0f / EMB) - mu * mu;
        const float sc  = rsqrtf(var + LN_EPS);
#pragma unroll
        for (int j = 0; j < 8; ++j) {
            const int col = lane * 8 + j;
            hrow[row][col] = (v[j] - mu) * sc * gamma[col] + beta[col];
        }
    }
    __syncthreads();

    // ---------------- Phase 2: gather decoder rows + dot with h -------------
    // Per block: 16 rows x 128 candidates. Each wave handles 16 candidates per
    // row (c = wave, wave+8, ...). The h row is hoisted into registers once per
    // row (16x fewer LDS reads than per-dot loads, and no LDS in the hot loop).
    for (int r = 0; r < 16; ++r) {
        const int b = m0 + r;
        const int* crow = cidx + (size_t)b * NC;

        float hreg[8];
#pragma unroll
        for (int j = 0; j < 8; ++j) hreg[j] = hrow[r][lane * 8 + j];

#pragma unroll 4
        for (int ci = 0; ci < 16; ++ci) {
            const int c   = wave + ci * 8;
            const int idx = crow[c];                      // uniform -> SMEM load

            // Prefetch the gather row 4 candidates ahead; lanes cover the full
            // 1 KB row at 32B granularity (emits global_prefetch_b8).
            if (ci + 4 < 16) {
                const int idxp = crow[wave + (ci + 4) * 8];
                __builtin_prefetch(table + (size_t)idxp * EMB + lane * 8, 0, 0);
            }

            const float* trow = table + (size_t)idx * EMB;
            const float4 t0 = *(const float4*)(trow + lane * 8);
            const float4 t1 = *(const float4*)(trow + lane * 8 + 4);

            float p = t0.x * hreg[0] + t0.y * hreg[1] + t0.z * hreg[2] + t0.w * hreg[3]
                    + t1.x * hreg[4] + t1.y * hreg[5] + t1.z * hreg[6] + t1.w * hreg[7];
            p = wave_reduce_sum(p);
            if (lane == 0)
                out[(size_t)b * NC + c] = p + ebias[idx];
        }
    }
}

extern "C" void kernel_launch(void* const* d_in, const int* in_sizes, int n_in,
                              void* d_out, int out_size, void* d_ws, size_t ws_size,
                              hipStream_t stream) {
    (void)in_sizes; (void)n_in; (void)out_size; (void)d_ws; (void)ws_size;
    const float* X      = (const float*)d_in[0];
    const float* W      = (const float*)d_in[1];
    const float* bdense = (const float*)d_in[2];
    const float* gamma  = (const float*)d_in[3];
    const float* beta   = (const float*)d_in[4];
    const float* table  = (const float*)d_in[5];
    const float* ebias  = (const float*)d_in[6];
    const int*   cidx   = (const int*)d_in[7];
    float* out = (float*)d_out;

    dim3 grid(NB / 16);   // 256 blocks, each: 16 batch rows
    dim3 block(256);      // 8 wave32s
    hipLaunchKernelGGL(entity_head_fused_kernel, grid, block, 0, stream,
                       X, W, bdense, gamma, beta, table, ebias, cidx, out);
}
